// DMN_21466246545689
// MI455X (gfx1250) — matrix operand order repaired and checked by reading, not compiled, not run on hardware
//
#include <hip/hip_runtime.h>
#include <math.h>

#define H    128
#define VSZ  50000
#define BB   128
#define T_C  50
#define T_I  32
#define T_Q  32
#define NSEQ (BB * T_C)   // 6400 fact sequences
#define G3   (3 * H)      // 384 (GRU gate width)
#define NDEC 8
#define EPIS 3

typedef _Float16 f16;
typedef __attribute__((ext_vector_type(16))) _Float16 v16h;
typedef __attribute__((ext_vector_type(8)))  _Float16 v8h;
typedef __attribute__((ext_vector_type(8)))  float    v8f;

// ---------------------------------------------------------------------------
// WMMA GEMM, 32x32 output per wave (2x2 register blocking):
//   C[M,N] (f32, row stride ldc) = A[M,K](f16, lda) @ W[N,K]^T (f16, ldw) + bias
// Per k-step: 2 A-fragment loads + 2 B-fragment loads feed 4 WMMAs
// (2x data reuse vs 1 tile/wave).  M must be a multiple of 32; N a multiple
// of 16 (odd trailing 16-tile handled by a wave-uniform clamp so EXEC stays
// all-ones through the WMMAs).
// Fragment layouts per CDNA5 ISA 7.12.2:
//   A 16x32: lane -> M=lane&15; elems 0..7 = K kb..kb+7, 8..15 = kb+16..kb+23
//            (kb = 0 lanes 0-15, 8 lanes 16-31)
//   B 32x16: lane -> N=lane&15; same K order as A (kb = 0 lanes 0-15, 8 lanes 16-31)
//   C: VGPR v -> M = v + (lane>=16 ? 8 : 0), N = lane&15
// ---------------------------------------------------------------------------
__device__ __forceinline__ v16h load_a_frag(const f16* Arow, int k0, int kbA) {
    v8h lo = *(const v8h*)(Arow + k0 + kbA);
    v8h hi = *(const v8h*)(Arow + k0 + kbA + 16);
    return __builtin_shufflevector(lo, hi, 0,1,2,3,4,5,6,7,8,9,10,11,12,13,14,15);
}

__global__ void k_gemm_wmma(const f16* __restrict__ A, int lda,
                            const f16* __restrict__ W, int ldw,
                            const float* __restrict__ bias,
                            float* __restrict__ C, long long ldc,
                            int M, int N, int K)
{
    const int n0   = blockIdx.x * 32;
    const int m0   = blockIdx.y * 32;
    const int lane = threadIdx.x & 31;
    const int hs   = lane >> 4;
    const int l16  = lane & 15;
    const bool hasN1 = (n0 + 16) < N;           // wave-uniform
    const int  n1    = hasN1 ? (n0 + 16) : n0;  // clamp: duplicate work, store skipped

    const f16* Arow0 = A + (size_t)(m0 + l16) * lda;
    const f16* Arow1 = Arow0 + (size_t)16 * lda;
    const f16* Wrow0 = W + (size_t)(n0 + l16) * ldw;
    const f16* Wrow1 = W + (size_t)(n1 + l16) * ldw;
    const int  kbA   = hs ? 8 : 0;
    const int  kbB   = hs ? 8 : 0;

    v8f acc00 = {}, acc01 = {}, acc10 = {}, acc11 = {};
    for (int k0 = 0; k0 < K; k0 += 32) {
        v16h a0 = load_a_frag(Arow0, k0, kbA);
        v16h a1 = load_a_frag(Arow1, k0, kbA);
        v16h b0 = load_a_frag(Wrow0, k0, kbB);
        v16h b1 = load_a_frag(Wrow1, k0, kbB);
        acc00 = __builtin_amdgcn_wmma_f32_16x16x32_f16(false, a0, false, b0, (short)0, acc00, false, false);
        acc01 = __builtin_amdgcn_wmma_f32_16x16x32_f16(false, a0, false, b1, (short)0, acc01, false, false);
        acc10 = __builtin_amdgcn_wmma_f32_16x16x32_f16(false, a1, false, b0, (short)0, acc10, false, false);
        acc11 = __builtin_amdgcn_wmma_f32_16x16x32_f16(false, a1, false, b1, (short)0, acc11, false, false);
    }

    const float bn0 = bias ? bias[n0 + l16] : 0.0f;
    const float bn1 = (bias && hasN1) ? bias[n0 + 16 + l16] : 0.0f;
    const int rb0 = m0 + hs * 8;
    const int rb1 = m0 + 16 + hs * 8;
    #pragma unroll
    for (int v = 0; v < 8; ++v) {
        C[(size_t)(rb0 + v) * ldc + n0 + l16] = acc00[v] + bn0;
        C[(size_t)(rb1 + v) * ldc + n0 + l16] = acc10[v] + bn0;
        if (hasN1) {
            C[(size_t)(rb0 + v) * ldc + n0 + 16 + l16] = acc01[v] + bn1;
            C[(size_t)(rb1 + v) * ldc + n0 + 16 + l16] = acc11[v] + bn1;
        }
    }
}

// ---------------------------------------------------------------------------
// Elementwise / helper kernels
// ---------------------------------------------------------------------------
__global__ void k_cvt16(const float* __restrict__ s, f16* __restrict__ d, int n) {
    int i = blockIdx.x * blockDim.x + threadIdx.x;
    if (i < n) d[i] = (f16)s[i];
}
__global__ void k_copy_cvt(const float* __restrict__ s, float* __restrict__ d32,
                           f16* __restrict__ d16, int n) {
    int i = blockIdx.x * blockDim.x + threadIdx.x;
    if (i < n) { float v = s[i]; d32[i] = v; d16[i] = (f16)v; }
}
__global__ void k_zero32(float* p, int n) {
    int i = blockIdx.x * blockDim.x + threadIdx.x;
    if (i < n) p[i] = 0.0f;
}
__global__ void k_zero16(f16* p, int n) {
    int i = blockIdx.x * blockDim.x + threadIdx.x;
    if (i < n) p[i] = (f16)0.0f;
}
__global__ void k_flen(const int* __restrict__ mask, int* __restrict__ flen,
                       int T, int M) {
    int m = blockIdx.x * blockDim.x + threadIdx.x;
    if (m >= M) return;
    int c = 0;
    for (int i = 0; i < T; ++i) c += (mask[(size_t)m * T + i] == 0);
    flen[m] = c;
}
// x16[m,:] = f16(embed[tokens[m*tstride + t], :])   grid=M, block=H
__global__ void k_gather_embed(const int* __restrict__ tok, int tstride, int t,
                               const float* __restrict__ embed,
                               f16* __restrict__ x16) {
    int m = blockIdx.x, j = threadIdx.x;
    int id = tok[(size_t)m * tstride + t];
    x16[(size_t)m * H + j] = (f16)embed[(size_t)id * H + j];
}
// Fused GRU gate math; optionally latch encoder output at t == flen[m]-1.
// grid = M, block = H.  In-place safe on hnew==hprev (same-thread RMW).
__global__ void k_gru_combine(const float* __restrict__ gi, const float* __restrict__ gh,
                              const float* __restrict__ bih, const float* __restrict__ bhh,
                              const float* __restrict__ hprev,
                              float* __restrict__ hnew, f16* __restrict__ hnew16,
                              const int* __restrict__ flen, int t,
                              float* enc32, f16* enc16)
{
    int m = blockIdx.x, j = threadIdx.x;
    size_t gb = (size_t)m * G3;
    float ir = gi[gb + j]         + bih[j];
    float iz = gi[gb + H + j]     + bih[H + j];
    float in = gi[gb + 2 * H + j] + bih[2 * H + j];
    float hr = gh[gb + j]         + bhh[j];
    float hz = gh[gb + H + j]     + bhh[H + j];
    float hn = gh[gb + 2 * H + j] + bhh[2 * H + j];
    float h  = hprev[(size_t)m * H + j];
    float r  = 1.0f / (1.0f + __expf(-(ir + hr)));
    float z  = 1.0f / (1.0f + __expf(-(iz + hz)));
    float n  = tanhf(in + r * hn);
    float o  = (1.0f - z) * n + z * h;
    hnew[(size_t)m * H + j]   = o;
    hnew16[(size_t)m * H + j] = (f16)o;
    if (enc32 && flen[m] - 1 == t) {
        enc32[(size_t)m * H + j] = o;
        enc16[(size_t)m * H + j] = (f16)o;
    }
}
// z = [ft*q, ft*m, |ft-q|, |ft-m|] for fact s; grid=B, block=H
__global__ void k_gate_feat(const float* __restrict__ enc32, int s,
                            const float* __restrict__ q, const float* __restrict__ mem,
                            f16* __restrict__ z16) {
    int b = blockIdx.x, j = threadIdx.x;
    float ft = enc32[((size_t)b * T_C + s) * H + j];
    float qv = q[(size_t)b * H + j];
    float mv = mem[(size_t)b * H + j];
    size_t zb = (size_t)b * 4 * H;
    z16[zb + j]         = (f16)(ft * qv);
    z16[zb + H + j]     = (f16)(ft * mv);
    z16[zb + 2 * H + j] = (f16)fabsf(ft - qv);
    z16[zb + 3 * H + j] = (f16)fabsf(ft - mv);
}
// g[b] = sigmoid( sum_j tanh(gact[b,j]) * W2[j] + b2 );  grid=B, block=H
__global__ void k_gate2(const float* __restrict__ gact, const float* __restrict__ W2,
                        const float* __restrict__ b2, float* __restrict__ g) {
    __shared__ float red[H];
    int b = blockIdx.x, j = threadIdx.x;
    red[j] = tanhf(gact[(size_t)b * H + j]) * W2[j];
    __syncthreads();
    for (int s = H / 2; s > 0; s >>= 1) {
        if (j < s) red[j] += red[j + s];
        __syncthreads();
    }
    if (j == 0) g[b] = 1.0f / (1.0f + __expf(-(red[0] + b2[0])));
}
// h = g*h2 + (1-g)*h ; also refresh f16 mirror.  grid=B, block=H
__global__ void k_apply_gate(const float* __restrict__ g, const float* __restrict__ h2,
                             float* __restrict__ h, f16* __restrict__ h16) {
    int b = blockIdx.x, j = threadIdx.x;
    float gv = g[b];
    size_t i = (size_t)b * H + j;
    float v = gv * h2[i] + (1.0f - gv) * h[i];
    h[i] = v; h16[i] = (f16)v;
}
// yq = [embed[2], q]  (B x 2H, f16); grid=B, block=2H
__global__ void k_build_yq(const float* __restrict__ embed, const float* __restrict__ q,
                           f16* __restrict__ yq) {
    int b = blockIdx.x, j = threadIdx.x;
    float v = (j < H) ? embed[2 * H + j] : q[(size_t)b * H + (j - H)];
    yq[(size_t)b * 2 * H + j] = (f16)v;
}
// In-place log_softmax over rows of length V; grid = #rows, block = 256
__global__ void k_log_softmax(float* __restrict__ out, int nV) {
    __shared__ float red[256];
    float* p = out + (size_t)blockIdx.x * nV;
    int tid = threadIdx.x;
    float mx = -INFINITY;
    for (int i = tid; i < nV; i += 256) mx = fmaxf(mx, p[i]);
    red[tid] = mx; __syncthreads();
    for (int s = 128; s > 0; s >>= 1) {
        if (tid < s) red[tid] = fmaxf(red[tid], red[tid + s]);
        __syncthreads();
    }
    mx = red[0]; __syncthreads();
    float sum = 0.0f;
    for (int i = tid; i < nV; i += 256) sum += __expf(p[i] - mx);
    red[tid] = sum; __syncthreads();
    for (int s = 128; s > 0; s >>= 1) {
        if (tid < s) red[tid] += red[tid + s];
        __syncthreads();
    }
    float lse = mx + __logf(red[0]);
    for (int i = tid; i < nV; i += 256) p[i] -= lse;
}

// ---------------------------------------------------------------------------
extern "C" void kernel_launch(void* const* d_in, const int* in_sizes, int n_in,
                              void* d_out, int out_size, void* d_ws, size_t ws_size,
                              hipStream_t stream)
{
    (void)in_sizes; (void)n_in; (void)out_size; (void)ws_size;

    // ---- inputs (setup_inputs dict order) ----
    const int*   facts      = (const int*)  d_in[0];
    const int*   fact_masks = (const int*)  d_in[1];
    const int*   questions  = (const int*)  d_in[2];
    const int*   q_masks    = (const int*)  d_in[3];
    /* d_in[4] = num_decode (==8, compiled in) */
    const float* embed      = (const float*)d_in[5];
    const float* ig_Wih = (const float*)d_in[6],  *ig_Whh = (const float*)d_in[7];
    const float* ig_bih = (const float*)d_in[8],  *ig_bhh = (const float*)d_in[9];
    const float* qg_Wih = (const float*)d_in[10], *qg_Whh = (const float*)d_in[11];
    const float* qg_bih = (const float*)d_in[12], *qg_bhh = (const float*)d_in[13];
    const float* at_Wih = (const float*)d_in[14], *at_Whh = (const float*)d_in[15];
    const float* at_bih = (const float*)d_in[16], *at_bhh = (const float*)d_in[17];
    const float* me_Wih = (const float*)d_in[18], *me_Whh = (const float*)d_in[19];
    const float* me_bih = (const float*)d_in[20], *me_bhh = (const float*)d_in[21];
    const float* an_Wih = (const float*)d_in[22], *an_Whh = (const float*)d_in[23];
    const float* an_bih = (const float*)d_in[24], *an_bhh = (const float*)d_in[25];
    const float* gate_W1 = (const float*)d_in[26], *gate_b1 = (const float*)d_in[27];
    const float* gate_W2 = (const float*)d_in[28], *gate_b2 = (const float*)d_in[29];
    const float* fc_W   = (const float*)d_in[30], *fc_b   = (const float*)d_in[31];
    float* out = (float*)d_out;

    // ---- workspace bump allocator (256B aligned) ----
    char*  base = (char*)d_ws;
    size_t off  = 0;
    auto alloc = [&](size_t bytes) -> void* {
        off = (off + 255) & ~(size_t)255;
        void* p = base + off;
        off += bytes;
        return p;
    };
    auto a16 = [&](size_t n) { return (f16*)  alloc(n * sizeof(f16));   };
    auto a32 = [&](size_t n) { return (float*)alloc(n * sizeof(float)); };
    auto aI  = [&](size_t n) { return (int*)  alloc(n * sizeof(int));   };

    // f16 weight mirrors
    f16* igWih16 = a16(G3 * H);     f16* igWhh16 = a16(G3 * H);
    f16* qgWih16 = a16(G3 * H);     f16* qgWhh16 = a16(G3 * H);
    f16* atWih16 = a16(G3 * H);     f16* atWhh16 = a16(G3 * H);
    f16* meWih16 = a16(G3 * H);     f16* meWhh16 = a16(G3 * H);
    f16* anWih16 = a16(G3 * 2 * H); f16* anWhh16 = a16(G3 * H);
    f16* gW1_16  = a16(H * 4 * H);
    f16* fcW16   = a16((size_t)VSZ * H);
    // activations
    f16*   x16   = a16((size_t)NSEQ * H);
    float* h32   = a32((size_t)NSEQ * H);   f16* h16  = a16((size_t)NSEQ * H);
    float* gi32  = a32((size_t)NSEQ * G3);
    float* gh32  = a32((size_t)NSEQ * G3);
    float* enc32 = a32((size_t)NSEQ * H);   f16* enc16 = a16((size_t)NSEQ * H);
    int*   flenF = aI(NSEQ);                int* flenQ = aI(BB);
    float* q32   = a32(BB * H);             f16* q16   = a16(BB * H);
    float* mem32 = a32(BB * H);             f16* mem16 = a16(BB * H);
    float* e32   = a32(BB * H);             f16* e16   = a16(BB * H);
    float* h2_32 = a32(BB * H);             f16* h2_16 = a16(BB * H);
    f16*   z16   = a16(BB * 4 * H);
    float* gact  = a32(BB * H);
    float* gvec  = a32(BB);
    f16*   yq16  = a16(BB * 2 * H);
    float* gidec = a32(BB * G3);
    float* hd32  = a32(BB * H);             f16* hd16  = a16(BB * H);

    auto cvt = [&](const float* s, f16* d, int n) {
        k_cvt16<<<(n + 255) / 256, 256, 0, stream>>>(s, d, n);
    };
    auto gemm = [&](const f16* A, int lda, const f16* W, int ldw, const float* bias,
                    float* C, long long ldc, int M, int N, int K) {
        dim3 grid((N + 31) / 32, M / 32);   // M always a multiple of 32 here
        k_gemm_wmma<<<grid, 32, 0, stream>>>(A, lda, W, ldw, bias, C, ldc, M, N, K);
    };

    // ---- weight conversion ----
    cvt(ig_Wih, igWih16, G3 * H);   cvt(ig_Whh, igWhh16, G3 * H);
    cvt(qg_Wih, qgWih16, G3 * H);   cvt(qg_Whh, qgWhh16, G3 * H);
    cvt(at_Wih, atWih16, G3 * H);   cvt(at_Whh, atWhh16, G3 * H);
    cvt(me_Wih, meWih16, G3 * H);   cvt(me_Whh, meWhh16, G3 * H);
    cvt(an_Wih, anWih16, G3 * 2 * H); cvt(an_Whh, anWhh16, G3 * H);
    cvt(gate_W1, gW1_16, H * 4 * H);
    cvt(fc_W, fcW16, VSZ * H);

    // ---- sequence lengths ----
    k_flen<<<(NSEQ + 127) / 128, 128, 0, stream>>>(fact_masks, flenF, T_I, NSEQ);
    k_flen<<<1, 128, 0, stream>>>(q_masks, flenQ, T_Q, BB);

    // ---- fact encoder GRU (M = 6400 per step) ----
    {
        int n = NSEQ * H;
        k_zero32<<<(n + 255) / 256, 256, 0, stream>>>(h32, n);
        k_zero16<<<(n + 255) / 256, 256, 0, stream>>>(h16, n);
    }
    for (int t = 0; t < T_I; ++t) {
        k_gather_embed<<<NSEQ, H, 0, stream>>>(facts, T_I, t, embed, x16);
        gemm(x16, H, igWih16, H, nullptr, gi32, G3, NSEQ, G3, H);
        gemm(h16, H, igWhh16, H, nullptr, gh32, G3, NSEQ, G3, H);
        k_gru_combine<<<NSEQ, H, 0, stream>>>(gi32, gh32, ig_bih, ig_bhh,
                                              h32, h32, h16, flenF, t, enc32, enc16);
    }

    // ---- question encoder GRU (M = 128 per step; reuse big buffers) ----
    {
        int n = BB * H;
        k_zero32<<<(n + 255) / 256, 256, 0, stream>>>(h32, n);
        k_zero16<<<(n + 255) / 256, 256, 0, stream>>>(h16, n);
    }
    for (int t = 0; t < T_Q; ++t) {
        k_gather_embed<<<BB, H, 0, stream>>>(questions, T_Q, t, embed, x16);
        gemm(x16, H, qgWih16, H, nullptr, gi32, G3, BB, G3, H);
        gemm(h16, H, qgWhh16, H, nullptr, gh32, G3, BB, G3, H);
        k_gru_combine<<<BB, H, 0, stream>>>(gi32, gh32, qg_bih, qg_bhh,
                                            h32, h32, h16, flenQ, t, q32, q16);
    }

    // ---- episodic memory ----
    k_copy_cvt<<<(BB * H + 255) / 256, 256, 0, stream>>>(q32, mem32, mem16, BB * H);
    for (int ep = 0; ep < EPIS; ++ep) {
        int n = BB * H;
        k_zero32<<<(n + 255) / 256, 256, 0, stream>>>(e32, n);
        k_zero16<<<(n + 255) / 256, 256, 0, stream>>>(e16, n);
        for (int s = 0; s < T_C; ++s) {
            k_gate_feat<<<BB, H, 0, stream>>>(enc32, s, q32, mem32, z16);
            gemm(z16, 4 * H, gW1_16, 4 * H, gate_b1, gact, H, BB, H, 4 * H);
            k_gate2<<<BB, H, 0, stream>>>(gact, gate_W2, gate_b2, gvec);
            gemm(enc16 + (size_t)s * H, T_C * H, atWih16, H, nullptr, gi32, G3, BB, G3, H);
            gemm(e16, H, atWhh16, H, nullptr, gh32, G3, BB, G3, H);
            k_gru_combine<<<BB, H, 0, stream>>>(gi32, gh32, at_bih, at_bhh,
                                                e32, h2_32, h2_16,
                                                nullptr, 0, nullptr, nullptr);
            k_apply_gate<<<BB, H, 0, stream>>>(gvec, h2_32, e32, e16);
        }
        gemm(e16, H, meWih16, H, nullptr, gi32, G3, BB, G3, H);
        gemm(mem16, H, meWhh16, H, nullptr, gh32, G3, BB, G3, H);
        k_gru_combine<<<BB, H, 0, stream>>>(gi32, gh32, me_bih, me_bhh,
                                            mem32, mem32, mem16,
                                            nullptr, 0, nullptr, nullptr);
    }

    // ---- answer decoder ----
    k_build_yq<<<BB, 2 * H, 0, stream>>>(embed, q32, yq16);
    gemm(yq16, 2 * H, anWih16, 2 * H, nullptr, gidec, G3, BB, G3, 2 * H);  // fixed input
    k_copy_cvt<<<(BB * H + 255) / 256, 256, 0, stream>>>(mem32, hd32, hd16, BB * H);
    for (int t = 0; t < NDEC; ++t) {
        gemm(hd16, H, anWhh16, H, nullptr, gh32, G3, BB, G3, H);
        k_gru_combine<<<BB, H, 0, stream>>>(gidec, gh32, an_bih, an_bhh,
                                            hd32, hd32, hd16,
                                            nullptr, 0, nullptr, nullptr);
        // logits for step t -> out rows b*NDEC + t   (base + t*V, row stride 8V)
        gemm(hd16, H, fcW16, H, fc_b, out + (size_t)t * VSZ,
             (long long)NDEC * VSZ, BB, VSZ, H);
    }
    k_log_softmax<<<BB * NDEC, 256, 0, stream>>>(out, VSZ);
}
